// Qwen3Attention_56865366999255
// MI455X (gfx1250) — compile-verified, matrix operand order checked by reading, not run on hardware
//
#include <hip/hip_runtime.h>

// ---------------------------------------------------------------------------
// Qwen3 attention block for gfx1250 (MI455X), wave32 + WMMA f16->f32 + TDM.
// Pipeline: f32->f16 weight cvt, RMSnorm+cvt, QKV WMMA GEMMs (TDM-staged,
// explicitly 2x-unrolled double buffering), per-head RMSnorm+RoPE,
// flash-attention (TDM K-tiles, WMMA scores + WMMA PV, online softmax),
// f32 output-projection WMMA GEMM. TDM issue is guarded by a *scalar*
// wave-0 branch (readfirstlane) since tensor ops ignore EXEC.
// ---------------------------------------------------------------------------

typedef _Float16 half_t;
typedef __attribute__((ext_vector_type(16))) _Float16 v16h;
typedef __attribute__((ext_vector_type(8)))  _Float16 v8h;
typedef __attribute__((ext_vector_type(8)))  float    v8f;
typedef __attribute__((ext_vector_type(4)))  unsigned int u32x4;
typedef __attribute__((ext_vector_type(4)))  int i32x4;
typedef __attribute__((ext_vector_type(8)))  int i32x8;

#define S_LEN 2048
#define HID   4096
#define NQ    32
#define NKV   8
#define DH    128

#if defined(__has_builtin)
#if __has_builtin(__builtin_amdgcn_tensor_load_to_lds)
#define HAVE_TDM 1
#endif
#endif
#ifndef HAVE_TDM
#define HAVE_TDM 0
#endif

#if __has_include(<hip/amd_detail/amd_gfx1250_TDM.h>)
#define TDM_6ARG 1
#else
#define TDM_6ARG 0
#endif

// Scalar (SGPR) wave-0 test: forces s_cmp + s_cbranch instead of EXEC mask.
__device__ __forceinline__ bool is_wave0() {
  return __builtin_amdgcn_readfirstlane((int)threadIdx.x) < 32;
}

#if HAVE_TDM
// Issue a TDM 2-D tile load (global -> LDS), one per calling wave.
// tile_x elems contiguous per row, tile_y rows, row stride in elems,
// LDS padding: after every (2<<pad_iv) DWORDs insert (pad_amt+1) DWORDs.
__device__ __forceinline__ void tdm_load_2d(
    const half_t* gsrc, half_t* lds_dst,
    unsigned tile_x, unsigned tile_y,
    unsigned long long stride_elems,
    unsigned pad_iv, unsigned pad_amt)
{
  unsigned long long ga = (unsigned long long)(uintptr_t)gsrc;
  unsigned lds = (unsigned)(uintptr_t)lds_dst;   // low 32 bits = LDS byte offset

  u32x4 g0;
  g0[0] = 1u;                                        // count=1, user mode
  g0[1] = lds;                                       // lds_addr [63:32]
  g0[2] = (unsigned)(ga & 0xffffffffu);              // global_addr lo
  g0[3] = (unsigned)((ga >> 32) & 0x1ffffffu)        // global_addr [56:32]
        | 0x80000000u;                               // type = 2 ("image")

  unsigned td0 = (unsigned)stride_elems;             // tensor_dim0 (no OOB use)
  unsigned td1 = 0x40000000u;                        // tensor_dim1 (huge)
  i32x8 g1;
  g1[0] = (int)((1u << 16)                           // data_size = 2 bytes
              | (1u << 20)                           // pad_enable
              | (pad_iv << 22) | (pad_amt << 25));
  g1[1] = (int)((td0 & 0xffffu) << 16);              // atomic_barrier=0 | td0 lo
  g1[2] = (int)(((td0 >> 16) & 0xffffu) | ((td1 & 0xffffu) << 16));
  g1[3] = (int)(((td1 >> 16) & 0xffffu) | (tile_x << 16));   // | tile_dim0
  g1[4] = (int)(tile_y & 0xffffu);                   // tile_dim1 | tile_dim2=0
  g1[5] = (int)(stride_elems & 0xffffffffu);         // tensor_dim0_stride lo
  g1[6] = (int)((stride_elems >> 32) & 0xffffu);     // stride hi | dim1_stride=0
  g1[7] = 0;

  i32x4 z4 = {0, 0, 0, 0};
#if TDM_6ARG
  i32x8 z8 = {0, 0, 0, 0, 0, 0, 0, 0};
  __builtin_amdgcn_tensor_load_to_lds(g0, g1, z4, z4, z8, 0);
#else
  __builtin_amdgcn_tensor_load_to_lds(g0, g1, z4, z4, 0);
#endif
}
#endif  // HAVE_TDM

// Load a 16x32 f16 A/B fragment (ISA 7.12.2 wave32 layout) from a row-major
// (row, K-contiguous) padded tile.
__device__ __forceinline__ v16h load_frag(const half_t* __restrict__ base, int stride) {
  const int lane = threadIdx.x & 31;
  const int r    = lane & 15;
  const int koff = (lane >> 4) << 3;  // 0 or 8
  const half_t* p = base + r * stride + koff;
  v8h lo = *(const v8h*)(p);
  v8h hi = *(const v8h*)(p + 16);
  v16h out;
#pragma unroll
  for (int i = 0; i < 8; ++i) { out[i] = lo[i]; out[i + 8] = hi[i]; }
  return out;
}

constexpr int G_BM = 128, G_BN = 128, G_BK = 32, G_LDT = G_BK + 8;

// One K-step of the block-tile GEMM: 2 A-frags x 4 B-frags -> 8 WMMA.
__device__ __forceinline__ void gemm_step(
    const half_t* At, const half_t* Bt, int wm, int wn, v8f (&acc)[2][4])
{
  v16h a0 = load_frag(At + (wm * 32 +  0) * G_LDT, G_LDT);
  v16h a1 = load_frag(At + (wm * 32 + 16) * G_LDT, G_LDT);
#pragma unroll
  for (int nt = 0; nt < 4; ++nt) {
    v16h b = load_frag(Bt + (wn * 64 + nt * 16) * G_LDT, G_LDT);
    acc[0][nt] = __builtin_amdgcn_wmma_f32_16x16x32_f16(
        false, a0, false, b, (short)0, acc[0][nt], false, false);
    acc[1][nt] = __builtin_amdgcn_wmma_f32_16x16x32_f16(
        false, a1, false, b, (short)0, acc[1][nt], false, false);
  }
}

// ---------------------------------------------------------------------------
// C(MxN) = A(MxK) * B(NxK)^T, A/B f16 row-major (K contiguous), OutT output.
// Block tile 128x128, K-step 32, 8 waves as 4(m) x 2(n), 32x64 per wave.
// TDM double buffering, manually unrolled 2x (requires K % 64 == 0).
// ---------------------------------------------------------------------------
template <typename OutT>
__global__ __launch_bounds__(256) void gemm_tn_wmma(
    const half_t* __restrict__ A, const half_t* __restrict__ B,
    OutT* __restrict__ C, int M, int N, int K)
{
  constexpr int BM = G_BM, BN = G_BN, BK = G_BK, LDT = G_LDT;
  __shared__ __align__(16) half_t Asb[2 * BM * LDT];
  __shared__ __align__(16) half_t Bsb[2 * BN * LDT];

  const int tid  = threadIdx.x;
  const int lane = tid & 31;
  const int w    = tid >> 5;
  const int wm   = w & 3;      // 0..3 -> m offset wm*32
  const int wn   = w >> 2;     // 0..1 -> n offset wn*64
  const int m0   = blockIdx.y * BM;
  const int n0   = blockIdx.x * BN;

  v8f acc[2][4] = {};
  const int nsteps = K / BK;   // even by construction (K % 64 == 0)

#if HAVE_TDM
  half_t* As0 = &Asb[0];           half_t* As1 = &Asb[BM * LDT];
  half_t* Bs0 = &Bsb[0];           half_t* Bs1 = &Bsb[BN * LDT];
  if (is_wave0()) {   // scalar branch: only wave 0 drives the TDM
    tdm_load_2d(&A[(size_t)m0 * K], As0, BK, BM, (unsigned long long)K, 3, 3);
    tdm_load_2d(&B[(size_t)n0 * K], Bs0, BK, BN, (unsigned long long)K, 3, 3);
  }
  for (int t = 0; t < nsteps; t += 2) {
    // ---- even step: consume buf0, prefetch t+1 into buf1 ----
    if (is_wave0()) {
      int kn = (t + 1) * BK;
      tdm_load_2d(&A[(size_t)m0 * K + kn], As1, BK, BM, (unsigned long long)K, 3, 3);
      tdm_load_2d(&B[(size_t)n0 * K + kn], Bs1, BK, BN, (unsigned long long)K, 3, 3);
      __builtin_amdgcn_s_wait_tensorcnt(2);   // oldest 2 (tile t) complete
    }
    __syncthreads();
    gemm_step(As0, Bs0, wm, wn, acc);
    __syncthreads();
    // ---- odd step: consume buf1, prefetch t+2 into buf0 ----
    if (is_wave0()) {
      if (t + 2 < nsteps) {
        int kn = (t + 2) * BK;
        tdm_load_2d(&A[(size_t)m0 * K + kn], As0, BK, BM, (unsigned long long)K, 3, 3);
        tdm_load_2d(&B[(size_t)n0 * K + kn], Bs0, BK, BN, (unsigned long long)K, 3, 3);
        __builtin_amdgcn_s_wait_tensorcnt(2);
      } else {
        __builtin_amdgcn_s_wait_tensorcnt(0);
      }
    }
    __syncthreads();
    gemm_step(As1, Bs1, wm, wn, acc);
    __syncthreads();
  }
#else
  for (int t = 0; t < nsteps; ++t) {
    int k0 = t * BK;
#pragma unroll
    for (int i = 0; i < 2; ++i) {   // A: 512 chunks of 8 halves
      int c = tid * 2 + i, row = c >> 2, cc = (c & 3) << 3;
      *(v8h*)(&Asb[row * LDT + cc]) = *(const v8h*)(&A[(size_t)(m0 + row) * K + k0 + cc]);
    }
#pragma unroll
    for (int i = 0; i < 2; ++i) {   // B: 512 chunks
      int c = tid * 2 + i, row = c >> 2, cc = (c & 3) << 3;
      *(v8h*)(&Bsb[row * LDT + cc]) = *(const v8h*)(&B[(size_t)(n0 + row) * K + k0 + cc]);
    }
    __syncthreads();
    gemm_step(&Asb[0], &Bsb[0], wm, wn, acc);
    __syncthreads();
  }
#endif

  const int r16 = lane & 15;
  const int hi8 = (lane >> 4) * 8;
#pragma unroll
  for (int mt = 0; mt < 2; ++mt)
#pragma unroll
    for (int nt = 0; nt < 4; ++nt)
#pragma unroll
      for (int r = 0; r < 8; ++r) {
        int row = m0 + wm * 32 + mt * 16 + r + hi8;
        int col = n0 + wn * 64 + nt * 16 + r16;
        C[(size_t)row * N + col] = (OutT)acc[mt][nt][r];
      }
}

// ---------------------------------------------------------------------------
__global__ __launch_bounds__(256) void rmsnorm_x_kernel(
    const float* __restrict__ x, const float* __restrict__ w,
    half_t* __restrict__ out)
{
  __shared__ float red[8];
  const int s   = blockIdx.x;
  const int tid = threadIdx.x;
  const float* xr = x + (size_t)s * HID;
  float ss = 0.f;
#pragma unroll
  for (int i = 0; i < HID / 256; ++i) {
    float v = xr[tid + i * 256];
    ss += v * v;
  }
#pragma unroll
  for (int m = 16; m >= 1; m >>= 1) ss += __shfl_xor(ss, m, 32);
  if ((tid & 31) == 0) red[tid >> 5] = ss;
  __syncthreads();
  float tot = 0.f;
#pragma unroll
  for (int i = 0; i < 8; ++i) tot += red[i];
  float r = rsqrtf(tot * (1.0f / HID));
  half_t* orow = out + (size_t)s * HID;
#pragma unroll
  for (int i = 0; i < HID / 256; ++i) {
    int idx = tid + i * 256;
    orow[idx] = (half_t)(xr[idx] * r * w[idx]);
  }
}

// ---------------------------------------------------------------------------
__global__ __launch_bounds__(128) void qknorm_rope_kernel(
    half_t* __restrict__ buf, const float* __restrict__ nw,
    const float* __restrict__ cosb, const float* __restrict__ sinb,
    int nheads)
{
  __shared__ float xn[DH];
  __shared__ float red[4];
  const int s = blockIdx.x;
  const int h = blockIdx.y;
  const int i = threadIdx.x;
  const int stride = nheads * DH;
  half_t* row = buf + (size_t)s * stride + h * DH;
  float v  = (float)row[i];
  float ss = v * v;
#pragma unroll
  for (int m = 16; m >= 1; m >>= 1) ss += __shfl_xor(ss, m, 32);
  if ((i & 31) == 0) red[i >> 5] = ss;
  __syncthreads();
  float tot = red[0] + red[1] + red[2] + red[3];
  float r = rsqrtf(tot * (1.0f / DH));
  float n = v * r * nw[i];
  xn[i] = n;
  __syncthreads();
  float rot = (i < DH / 2) ? -xn[i + DH / 2] : xn[i - DH / 2];
  float c  = cosb[(size_t)s * DH + i];
  float sn = sinb[(size_t)s * DH + i];
  row[i] = (half_t)(n * c + rot * sn);
}

__global__ __launch_bounds__(256) void cvt_f32_f16_kernel(
    const float* __restrict__ src, half_t* __restrict__ dst, int n)
{
  int idx    = blockIdx.x * 256 + threadIdx.x;
  int stride = gridDim.x * 256;
  for (int i = idx; i < n; i += stride) dst[i] = (half_t)src[i];
}

// ---------------------------------------------------------------------------
// Causal GQA flash attention. Grid = (S/128, HQ). 8 waves, 16 q rows each.
// K-tile (32x128) staged by TDM (wave-0 scalar branch); V staged manually
// (needs transpose TDM cannot do).
// ---------------------------------------------------------------------------
__global__ __launch_bounds__(256) void attn_kernel(
    const half_t* __restrict__ q16, const half_t* __restrict__ k16,
    const half_t* __restrict__ v16, half_t* __restrict__ o16)
{
  constexpr int KT  = 32;
  constexpr int KLD = DH + 8;   // 136 halves (matches TDM pad 256B+16B)
  constexpr int VLD = KT + 8;   // 40 halves
  constexpr int PLD = KT + 8;
  __shared__ __align__(16) half_t Ks[KT * KLD];
  __shared__ __align__(16) half_t Vt[DH * VLD];
  __shared__ __align__(16) half_t Ps[8 * 16 * PLD];

  const int tid  = threadIdx.x;
  const int lane = tid & 31;
  const int w    = tid >> 5;
  const int head = blockIdx.y;
  const int kvh  = head >> 2;                // HQ/HKV = 4
  const int q0   = blockIdx.x * 128;
  const int qb   = q0 + w * 16;

  const int r16 = lane & 15;
  const int hi  = lane >> 4;

  v16h qf[4];
#pragma unroll
  for (int c = 0; c < 4; ++c)
    qf[c] = load_frag(q16 + (size_t)qb * (NQ * DH) + head * DH + c * 32, NQ * DH);

  v8f o[8] = {};
  float mrow[8], lrow[8];
#pragma unroll
  for (int r = 0; r < 8; ++r) { mrow[r] = -1e30f; lrow[r] = 0.f; }

  const float scale = 0.08838834764831845f;  // 1/sqrt(128)
  const int kend = q0 + 128;

  for (int k0 = 0; k0 < kend; k0 += KT) {
#if HAVE_TDM
    if (is_wave0())   // K tile: 32 rows x 128 halves, stride 1024, pad 64DW+4DW
      tdm_load_2d(k16 + (size_t)k0 * (NKV * DH) + kvh * DH, Ks,
                  DH, KT, (unsigned long long)(NKV * DH), 5, 3);
#endif
#pragma unroll
    for (int i = 0; i < 2; ++i) {    // V tile load + transpose into Vt
      int c   = tid * 2 + i;         // 0..511
      int row = c >> 4;              // 0..31
      int cc  = (c & 15) << 3;       // 0..120
      size_t g = (size_t)(k0 + row) * (NKV * DH) + kvh * DH + cc;
#if !HAVE_TDM
      *(v8h*)(&Ks[row * KLD + cc]) = *(const v8h*)(&k16[g]);
#endif
      v8h vv = *(const v8h*)(&v16[g]);
#pragma unroll
      for (int j = 0; j < 8; ++j) Vt[(cc + j) * VLD + row] = vv[j];
    }
#if HAVE_TDM
    if (is_wave0()) __builtin_amdgcn_s_wait_tensorcnt(0);
#endif
    __syncthreads();

    if (k0 <= qb + 15) {
      v8f sf[2] = {};
#pragma unroll
      for (int nt = 0; nt < 2; ++nt) {
#pragma unroll
        for (int c = 0; c < 4; ++c) {
          v16h bf = load_frag(&Ks[(nt * 16) * KLD + c * 32], KLD);
          sf[nt] = __builtin_amdgcn_wmma_f32_16x16x32_f16(
              false, qf[c], false, bf, (short)0, sf[nt], false, false);
        }
      }
#pragma unroll
      for (int nt = 0; nt < 2; ++nt) {
        int kcol = k0 + nt * 16 + r16;
#pragma unroll
        for (int r = 0; r < 8; ++r) {
          int qrow = qb + r + hi * 8;
          float sv = sf[nt][r] * scale;
          sf[nt][r] = (kcol <= qrow) ? sv : -3e30f;
        }
      }
      half_t* myP = &Ps[w * 16 * PLD];
#pragma unroll
      for (int r = 0; r < 8; ++r) {
        float vmax = fmaxf(sf[0][r], sf[1][r]);
#pragma unroll
        for (int m = 8; m >= 1; m >>= 1) vmax = fmaxf(vmax, __shfl_xor(vmax, m, 16));
        float mnew  = fmaxf(mrow[r], vmax);
        float alpha = __expf(mrow[r] - mnew);
        float p0 = __expf(sf[0][r] - mnew);
        float p1 = __expf(sf[1][r] - mnew);
        float psum = p0 + p1;
#pragma unroll
        for (int m = 8; m >= 1; m >>= 1) psum += __shfl_xor(psum, m, 16);
        lrow[r] = lrow[r] * alpha + psum;
        mrow[r] = mnew;
#pragma unroll
        for (int n = 0; n < 8; ++n) o[n][r] *= alpha;
        myP[(r + hi * 8) * PLD + r16]      = (half_t)p0;
        myP[(r + hi * 8) * PLD + 16 + r16] = (half_t)p1;
      }
      v16h pa = load_frag(myP, PLD);
#pragma unroll
      for (int n = 0; n < 8; ++n) {
        v16h bv = load_frag(&Vt[(n * 16) * VLD], VLD);
        o[n] = __builtin_amdgcn_wmma_f32_16x16x32_f16(
            false, pa, false, bv, (short)0, o[n], false, false);
      }
    }
    __syncthreads();
  }

#pragma unroll
  for (int r = 0; r < 8; ++r) {
    float inv = 1.0f / lrow[r];
    int qrow = qb + r + hi * 8;
#pragma unroll
    for (int n = 0; n < 8; ++n)
      o16[(size_t)qrow * (NQ * DH) + head * DH + n * 16 + r16] =
          (half_t)(o[n][r] * inv);
  }
}

// ---------------------------------------------------------------------------
extern "C" void kernel_launch(void* const* d_in, const int* in_sizes, int n_in,
                              void* d_out, int out_size, void* d_ws, size_t ws_size,
                              hipStream_t stream) {
  (void)in_sizes; (void)n_in; (void)out_size; (void)ws_size;
  const float* x    = (const float*)d_in[0];
  const float* cosb = (const float*)d_in[1];
  const float* sinb = (const float*)d_in[2];
  const float* ln_w = (const float*)d_in[3];
  const float* qnw  = (const float*)d_in[4];
  const float* knw  = (const float*)d_in[5];
  const float* wq   = (const float*)d_in[6];
  const float* wk   = (const float*)d_in[7];
  const float* wv   = (const float*)d_in[8];
  const float* wo   = (const float*)d_in[9];
  float* out = (float*)d_out;

  char* ws = (char*)d_ws;
  size_t off = 0;
  auto alloc = [&](size_t bytes) -> char* {
    char* p = ws + off;
    off += (bytes + 255) & ~(size_t)255;
    return p;
  };
  half_t* xn16 = (half_t*)alloc((size_t)S_LEN * HID * 2);   // later: attn out
  half_t* wbig = (half_t*)alloc((size_t)HID * HID * 2);     // wq16, later wo16
  half_t* wk16 = (half_t*)alloc((size_t)NKV * DH * HID * 2);
  half_t* wv16 = (half_t*)alloc((size_t)NKV * DH * HID * 2);
  half_t* q16  = (half_t*)alloc((size_t)S_LEN * NQ * DH * 2);
  half_t* k16  = (half_t*)alloc((size_t)S_LEN * NKV * DH * 2);
  half_t* v16  = (half_t*)alloc((size_t)S_LEN * NKV * DH * 2);
  half_t* a16  = xn16;

  cvt_f32_f16_kernel<<<4096, 256, 0, stream>>>(wq, wbig, HID * HID);
  cvt_f32_f16_kernel<<<2048, 256, 0, stream>>>(wk, wk16, NKV * DH * HID);
  cvt_f32_f16_kernel<<<2048, 256, 0, stream>>>(wv, wv16, NKV * DH * HID);

  rmsnorm_x_kernel<<<S_LEN, 256, 0, stream>>>(x, ln_w, xn16);

  gemm_tn_wmma<half_t><<<dim3(HID / 128, S_LEN / 128), 256, 0, stream>>>(
      xn16, wbig, q16, S_LEN, HID, HID);
  gemm_tn_wmma<half_t><<<dim3((NKV * DH) / 128, S_LEN / 128), 256, 0, stream>>>(
      xn16, wk16, k16, S_LEN, NKV * DH, HID);
  gemm_tn_wmma<half_t><<<dim3((NKV * DH) / 128, S_LEN / 128), 256, 0, stream>>>(
      xn16, wv16, v16, S_LEN, NKV * DH, HID);

  qknorm_rope_kernel<<<dim3(S_LEN, NQ),  128, 0, stream>>>(q16, qnw, cosb, sinb, NQ);
  qknorm_rope_kernel<<<dim3(S_LEN, NKV), 128, 0, stream>>>(k16, knw, cosb, sinb, NKV);

  attn_kernel<<<dim3(S_LEN / 128, NQ), 256, 0, stream>>>(q16, k16, v16, a16);

  cvt_f32_f16_kernel<<<4096, 256, 0, stream>>>(wo, wbig, HID * HID);
  gemm_tn_wmma<float><<<dim3(HID / 128, S_LEN / 128), 256, 0, stream>>>(
      a16, wbig, out, S_LEN, HID, HID);
}
